// Model_30872224924053
// MI455X (gfx1250) — matrix operand order repaired and checked by reading, not compiled, not run on hardware
//
#include <hip/hip_runtime.h>

// ---------------------------------------------------------------------------
// MI455X (gfx1250, wave32) implementation.
//   K1: embedding gather -> bf16 E[B][S][E]
//   K0a/b/c: pack Wc = conv_w @ w_ih^T (768x1024) and W_hh into WMMA B-operand
//            fragment layout (bf16), fold biases: gb = b_ih+b_hh+conv_b@w_ih^T
//   K2: G[t][b][4H] = im2col(E) @ Wc + gb   -- big bf16 WMMA GEMM, stores raw
//       16x16 C fragments (bf16) so the recurrence can reload them directly.
//   K3: single-workgroup (32 waves) LSTM recurrence using
//       v_wmma_f32_16x16x32_bf16, h/c/hsum in LDS, barrier per step; masked
//       mean + linear + sigmoid fused at the end.
// ---------------------------------------------------------------------------

#define BATCH   64
#define SEQ     1024
#define EMB     256
#define HID     256
#define TSTEPS  1022          // SEQ - 3 + 1
#define KCONV   768           // 3*EMB
#define GATES   1024          // 4*HID

typedef __attribute__((ext_vector_type(16))) __bf16 v16bf;
typedef __attribute__((ext_vector_type(8)))  float  v8f;

__device__ __forceinline__ unsigned short f2bf(float x) {
  unsigned u = __float_as_uint(x);
  return (unsigned short)((u + 0x7FFFu + ((u >> 16) & 1u)) >> 16);  // RNE
}

__device__ __forceinline__ v8f expand8(uint4 q) {   // 8 packed bf16 -> 8 f32
  v8f r;
  r[0] = __uint_as_float(q.x << 16); r[1] = __uint_as_float(q.x & 0xFFFF0000u);
  r[2] = __uint_as_float(q.y << 16); r[3] = __uint_as_float(q.y & 0xFFFF0000u);
  r[4] = __uint_as_float(q.z << 16); r[5] = __uint_as_float(q.z & 0xFFFF0000u);
  r[6] = __uint_as_float(q.w << 16); r[7] = __uint_as_float(q.w & 0xFFFF0000u);
  return r;
}

__device__ __forceinline__ uint4 pack8(v8f a) {     // 8 f32 -> 8 packed bf16
  uint4 o;
  o.x = (unsigned)f2bf(a[0]) | ((unsigned)f2bf(a[1]) << 16);
  o.y = (unsigned)f2bf(a[2]) | ((unsigned)f2bf(a[3]) << 16);
  o.z = (unsigned)f2bf(a[4]) | ((unsigned)f2bf(a[5]) << 16);
  o.w = (unsigned)f2bf(a[6]) | ((unsigned)f2bf(a[7]) << 16);
  return o;
}

// B fragment: 32 contiguous bytes per lane from the packed fragment buffer.
__device__ __forceinline__ v16bf load_fragB(const unsigned short* p) {
  union { v16bf v; uint4 q[2]; } u;
  u.q[0] = *(const uint4*)(p);
  u.q[1] = *(const uint4*)(p + 8);
  return u.v;
}
// A fragment from row-major bf16: 32 contiguous bytes per lane (k = 16*half + s).
__device__ __forceinline__ v16bf load_fragA(const unsigned short* p) {
  union { v16bf v; uint4 q[2]; } u;
  u.q[0] = *(const uint4*)(p);
  u.q[1] = *(const uint4*)(p + 8);
  return u.v;
}

__device__ __forceinline__ v8f wmma_bf16(v16bf a, v16bf b, v8f c) {
  return __builtin_amdgcn_wmma_f32_16x16x32_bf16(false, a, false, b,
                                                 (short)0, c, false, false);
}

__device__ __forceinline__ float sigmoidf(float x) { return 1.f / (1.f + __expf(-x)); }

// ---------------- K1: embedding gather to bf16 -----------------------------
__global__ void __launch_bounds__(256) k_embed(const int* __restrict__ ipts,
                                               const float* __restrict__ emb,
                                               unsigned short* __restrict__ E) {
  int row = blockIdx.x;                 // (b,s) flat, 65536 rows
  int e   = threadIdx.x;
  int tok = ipts[row];
  E[row * EMB + e] = f2bf(emb[tok * EMB + e]);
}

// ---------------- K0a: Wc = conv_w @ w_ih^T, packed bf16 B fragments -------
__global__ void __launch_bounds__(256) k_pack_wc(const float* __restrict__ conv_w,
                                                 const float* __restrict__ w_ih,
                                                 unsigned short* __restrict__ WcP) {
  int id = blockIdx.x * 256 + threadIdx.x;     // 768*1024
  int kk = id >> 10, g = id & 1023;
  const float* cw = conv_w + kk * 256;         // conv_w[w][e][f], kk=w*256+e
  const float* wi = w_ih + g * 256;
  float s = 0.f;
  for (int f = 0; f < 256; ++f) s += cw[f] * wi[f];
  int kt = kk >> 5, kr = kk & 31, nt = g >> 4, nr = g & 15;
  int lane = nr + ((kr >> 4) << 4), pos = kr & 15;
  WcP[(((nt * 24) + kt) * 32 + lane) * 16 + pos] = f2bf(s);
}

// ---------------- K0b: pack W_hh into bf16 B fragments ---------------------
__global__ void __launch_bounds__(256) k_pack_whh(const float* __restrict__ w_hh,
                                                  unsigned short* __restrict__ WhhP) {
  int id = blockIdx.x * 256 + threadIdx.x;     // 1024*256
  int g = id >> 8, j = id & 255;               // B[k=j][n=g] = w_hh[g][j]
  int kt = j >> 5, kr = j & 31, nt = g >> 4, nr = g & 15;
  int lane = nr + ((kr >> 4) << 4), pos = kr & 15;
  WhhP[(((nt * 8) + kt) * 32 + lane) * 16 + pos] = f2bf(w_hh[g * 256 + j]);
}

// ---------------- K0c: fused gate bias -------------------------------------
__global__ void __launch_bounds__(256) k_gb(const float* __restrict__ conv_b,
                                            const float* __restrict__ w_ih,
                                            const float* __restrict__ b_ih,
                                            const float* __restrict__ b_hh,
                                            float* __restrict__ gb) {
  int g = blockIdx.x * 256 + threadIdx.x;      // 1024
  float s = b_ih[g] + b_hh[g];
  const float* wi = w_ih + g * 256;
  for (int f = 0; f < 256; ++f) s += conv_b[f] * wi[f];
  gb[g] = s;
}

// ---------------- K2: gate pre-activations, big WMMA GEMM ------------------
// grid (TSTEPS, 4 mtiles), 256 threads = 8 waves, each wave 8 N-tiles.
__global__ void __launch_bounds__(256) k_gates(const unsigned short* __restrict__ E,
                                               const unsigned short* __restrict__ WcP,
                                               const float* __restrict__ gb,
                                               unsigned short* __restrict__ G) {
  int t = blockIdx.x, mt = blockIdx.y;
  int wave = threadIdx.x >> 5, lane = threadIdx.x & 31;
  int hs = lane >> 4, lr = lane & 15;

  v8f acc[8];
#pragma unroll
  for (int i = 0; i < 8; ++i) {
    float bv = gb[(wave * 8 + i) * 16 + lr];   // same N column for all 8 rows
    acc[i] = (v8f){bv, bv, bv, bv, bv, bv, bv, bv};
  }

  for (int kt = 0; kt < 24; ++kt) {            // K = 768
    int wwin = kt >> 3;                        // conv tap 0..2
    int e0   = (kt & 7) * 32 + hs * 16;
    int b    = mt * 16 + lr;
    const unsigned short* ap = E + ((b * SEQ) + t + wwin) * EMB + e0;
    v16bf A = load_fragA(ap);
#pragma unroll
    for (int i = 0; i < 8; ++i) {
      int n = wave * 8 + i;
      v16bf B = load_fragB(WcP + ((n * 24 + kt) * 32 + lane) * 16);
      acc[i] = wmma_bf16(A, B, acc[i]);
    }
  }
#pragma unroll
  for (int i = 0; i < 8; ++i) {
    int n = wave * 8 + i;                      // store raw bf16 C fragment
    *(uint4*)(G + (((t * 4 + mt) * 64 + n) * 32 + lane) * 8) = pack8(acc[i]);
  }
}

// ---------------- K3: sequential LSTM recurrence (1 WG, 32 waves) ----------
__global__ void __launch_bounds__(1024) k_lstm(const unsigned short* __restrict__ G,
                                               const unsigned short* __restrict__ WhhP,
                                               const int* __restrict__ seq_lengths,
                                               const float* __restrict__ h0,
                                               const float* __restrict__ c0,
                                               const float* __restrict__ lin_w,
                                               const float* __restrict__ lin_b,
                                               float* __restrict__ out) {
  extern __shared__ char smem[];
  unsigned short* h_lds = (unsigned short*)smem;            // 64*256 bf16 = 32KB
  float* c_lds = (float*)(smem + 32768);                    // 64KB
  float* hsum  = (float*)(smem + 32768 + 65536);            // 64KB

  int tid = threadIdx.x;
  for (int i = tid; i < BATCH * HID; i += 1024) {
    h_lds[i] = f2bf(h0[i]);
    c_lds[i] = c0[i];
    hsum[i]  = 0.f;
  }
  __syncthreads();

  int wave = tid >> 5, lane = tid & 31;
  int mt = wave >> 3, wq = wave & 7;           // 4 M-tiles x 8 waves
  int hs = lane >> 4, lr = lane & 15;

  int sl[8];
#pragma unroll
  for (int r = 0; r < 8; ++r) sl[r] = seq_lengths[mt * 16 + r + hs * 8];

  for (int t = 0; t < TSTEPS; ++t) {
    float hnew[2][8];
    if (t + 1 < TSTEPS)                        // warm L2/WGP$ for next slab
      __builtin_prefetch(G + (((t + 1) * 4 + mt) * 64) * 32 * 8, 0, 0);

#pragma unroll
    for (int qi = 0; qi < 2; ++qi) {
      int qg = wq * 2 + qi;                    // hidden-column tile 0..15
      const unsigned short* gbase = G + ((t * 4 + mt) * 64) * 256;
      v8f ai = expand8(*(const uint4*)(gbase + ((0 * 16 + qg) * 32 + lane) * 8));
      v8f af = expand8(*(const uint4*)(gbase + ((1 * 16 + qg) * 32 + lane) * 8));
      v8f ag = expand8(*(const uint4*)(gbase + ((2 * 16 + qg) * 32 + lane) * 8));
      v8f ao = expand8(*(const uint4*)(gbase + ((3 * 16 + qg) * 32 + lane) * 8));

#pragma unroll
      for (int kt = 0; kt < 8; ++kt) {         // K = 256 over h
        const unsigned short* ap = h_lds + (mt * 16 + lr) * HID + kt * 32 + hs * 16;
        v16bf A = load_fragA(ap);
        ai = wmma_bf16(A, load_fragB(WhhP + (((0 * 16 + qg) * 8 + kt) * 32 + lane) * 16), ai);
        af = wmma_bf16(A, load_fragB(WhhP + (((1 * 16 + qg) * 8 + kt) * 32 + lane) * 16), af);
        ag = wmma_bf16(A, load_fragB(WhhP + (((2 * 16 + qg) * 8 + kt) * 32 + lane) * 16), ag);
        ao = wmma_bf16(A, load_fragB(WhhP + (((3 * 16 + qg) * 8 + kt) * 32 + lane) * 16), ao);
      }
#pragma unroll
      for (int r = 0; r < 8; ++r) {            // C layout: b = mt*16 + r + 8*hs
        int b = mt * 16 + r + hs * 8;
        int j = qg * 16 + lr;
        float cc = sigmoidf(af[r]) * c_lds[b * HID + j] +
                   sigmoidf(ai[r]) * tanhf(ag[r]);
        c_lds[b * HID + j] = cc;
        float hh = sigmoidf(ao[r]) * tanhf(cc);
        if (t < sl[r]) hsum[b * HID + j] += hh;
        hnew[qi][r] = hh;
      }
    }
    __syncthreads();                           // all waves done reading old h
#pragma unroll
    for (int qi = 0; qi < 2; ++qi) {
      int qg = wq * 2 + qi;
#pragma unroll
      for (int r = 0; r < 8; ++r)
        h_lds[(mt * 16 + r + hs * 8) * HID + qg * 16 + lr] = f2bf(hnew[qi][r]);
    }
    __syncthreads();                           // new h visible for next step
  }

  if (tid < BATCH) {                           // masked mean + linear + sigmoid
    float inv = 1.f / (float)seq_lengths[tid];
    float s = 0.f;
    for (int j = 0; j < HID; ++j) s += hsum[tid * HID + j] * lin_w[j];
    out[tid] = sigmoidf(s * inv + lin_b[0]);
  }
}

// ---------------------------------------------------------------------------
extern "C" void kernel_launch(void* const* d_in, const int* in_sizes, int n_in,
                              void* d_out, int out_size, void* d_ws, size_t ws_size,
                              hipStream_t stream) {
  const int*   ipts   = (const int*)  d_in[0];
  const int*   seqlen = (const int*)  d_in[1];
  const float* h0     = (const float*)d_in[2];
  const float* c0     = (const float*)d_in[3];
  const float* emb    = (const float*)d_in[4];
  const float* conv_w = (const float*)d_in[5];
  const float* conv_b = (const float*)d_in[6];
  const float* w_ih   = (const float*)d_in[7];
  const float* w_hh   = (const float*)d_in[8];
  const float* b_ih   = (const float*)d_in[9];
  const float* b_hh   = (const float*)d_in[10];
  const float* lin_w  = (const float*)d_in[11];
  const float* lin_b  = (const float*)d_in[12];
  float* out = (float*)d_out;

  char* ws = (char*)d_ws;                                  // workspace layout
  unsigned short* E    = (unsigned short*)(ws);            // 33,554,432 B
  unsigned short* WcP  = (unsigned short*)(ws + 33554432); //  1,572,864 B
  unsigned short* WhhP = (unsigned short*)(ws + 35127296); //    524,288 B
  float*          gb   = (float*)        (ws + 35651584);  //      4,096 B
  unsigned short* G    = (unsigned short*)(ws + 35655680); // 133,955,584 B

  k_embed   <<<BATCH * SEQ, 256, 0, stream>>>(ipts, emb, E);
  k_pack_wc <<<(KCONV * GATES) / 256, 256, 0, stream>>>(conv_w, w_ih, WcP);
  k_pack_whh<<<(GATES * HID) / 256, 256, 0, stream>>>(w_hh, WhhP);
  k_gb      <<<GATES / 256, 256, 0, stream>>>(conv_b, w_ih, b_ih, b_hh, gb);
  k_gates   <<<dim3(TSTEPS, 4), 256, 0, stream>>>(E, WcP, gb, G);
  k_lstm    <<<1, 1024, 163840, stream>>>(G, WhhP, seqlen, h0, c0, lin_w, lin_b, out);
}
